// K2Layer_4234837753981
// MI455X (gfx1250) — matrix operand order repaired and checked by reading, not compiled, not run on hardware
//
#include <hip/hip_runtime.h>
#include <math.h>

// ---------------------------------------------------------------------------
// K2Layer for MI455X (gfx1250, wave32, WMMA).
// Key algebraic fusion: the rank-R decayed retention collapses into a single
// W x W score matrix S[b] = gate*kb + alpha * (q_i . diag(gamma^(i-j)) . k_j),
// so the layer becomes 4 dense GEMMs done with v_wmma_f32_16x16x32_bf16.
// ---------------------------------------------------------------------------

typedef __bf16 bf16;
typedef __attribute__((ext_vector_type(16))) __bf16 v16bf;
typedef __attribute__((ext_vector_type(8)))  float  v8f;

__device__ __forceinline__ float sigmoid_f(float x) { return 1.0f / (1.0f + __expf(-x)); }

// ---------------------------------------------------------------------------
// RMSNorm: one block (256 thr) per row, D == 1024 (4 floats per thread).
// ---------------------------------------------------------------------------
template <bool BF16OUT>
__global__ __launch_bounds__(256) void rmsnorm_kernel(const float* __restrict__ x,
                                                      const float* __restrict__ scale,
                                                      void* __restrict__ yv, int D) {
  const int row = blockIdx.x;
  const int t = threadIdx.x;
  const float* xr = x + (size_t)row * D;
  float4 xv = ((const float4*)xr)[t];
  float s = xv.x * xv.x + xv.y * xv.y + xv.z * xv.z + xv.w * xv.w;
  __shared__ float red[256];
  red[t] = s;
  __syncthreads();
  for (int off = 128; off > 0; off >>= 1) {
    if (t < off) red[t] += red[t + off];
    __syncthreads();
  }
  const float inv = rsqrtf(red[0] / (float)D + 1e-8f);
  float4 sv = ((const float4*)scale)[t];
  float o0 = xv.x * inv * sv.x, o1 = xv.y * inv * sv.y;
  float o2 = xv.z * inv * sv.z, o3 = xv.w * inv * sv.w;
  if (BF16OUT) {
    bf16* y = (bf16*)yv + (size_t)row * D + 4 * t;
    y[0] = (bf16)o0; y[1] = (bf16)o1; y[2] = (bf16)o2; y[3] = (bf16)o3;
  } else {
    float4 o; o.x = o0; o.y = o1; o.z = o2; o.w = o3;
    ((float4*)((float*)yv + (size_t)row * D))[t] = o;
  }
}

// ---------------------------------------------------------------------------
// q = l2norm(hn @ u), k = l2norm(hn @ v). One block per (b,w) row. D=1024,R=16.
// ---------------------------------------------------------------------------
__global__ __launch_bounds__(256) void qk_kernel(const float* __restrict__ hn,
                                                 const float* __restrict__ u,
                                                 const float* __restrict__ v,
                                                 float* __restrict__ q,
                                                 float* __restrict__ k, int D, int R) {
  const int row = blockIdx.x;
  const int t = threadIdx.x;
  __shared__ float sh[1024];
  __shared__ float part[32][9];
  __shared__ float dots[32];
  const float* hr = hn + (size_t)row * D;
  ((float4*)sh)[t] = ((const float4*)hr)[t];
  __syncthreads();
  const int o = t >> 3, li = t & 7;
  const float* Wm = (o < 16) ? u : v;
  const int c = (o < 16) ? o : (o - 16);
  float a = 0.f;
  for (int d = li; d < D; d += 8) a += sh[d] * Wm[(size_t)d * R + c];
  part[o][li] = a;
  __syncthreads();
  if (t < 32) {
    float s = 0.f;
    for (int e = 0; e < 8; e++) s += part[t][e];
    dots[t] = s;
  }
  __syncthreads();
  if (t < 32) {
    const int base = (t < 16) ? 0 : 16;
    float nn = 0.f;
    for (int e = 0; e < 16; e++) { float d0 = dots[base + e]; nn += d0 * d0; }
    const float val = dots[t] / fmaxf(sqrtf(nn), 1e-8f);
    if (t < 16) q[(size_t)row * 16 + t] = val;
    else        k[(size_t)row * 16 + (t - 16)] = val;
  }
}

// ---------------------------------------------------------------------------
// pw[r*W + d] = gamma_r^d, gamma_r = 0.85 + 0.15*sigmoid(decay_logit[r]).
// ---------------------------------------------------------------------------
__global__ void powtab_kernel(const float* __restrict__ dl, float* __restrict__ pw,
                              int W, int R) {
  const int i = blockIdx.x * blockDim.x + threadIdx.x;
  if (i >= W * R) return;
  const int r = i / W, d = i % W;
  const float g = 0.85f + 0.15f * sigmoid_f(dl[r]);
  pw[i] = __expf((float)d * __logf(fmaxf(g, 1e-8f)));
}

// ---------------------------------------------------------------------------
// S[b,i,j] = causal * ( gate*kb[i,j] + alpha * sum_r q[b,i,r]*gamma_r^(i-j)*k[b,j,r] )
// stored bf16 row-major (A-operand of gemm1). One block per (i, b).
// ---------------------------------------------------------------------------
__global__ __launch_bounds__(256) void buildS_kernel(const float* __restrict__ q,
                                                     const float* __restrict__ kk,
                                                     const float* __restrict__ pw,
                                                     const float* __restrict__ kbase,
                                                     const float* __restrict__ gate_logit,
                                                     const float* __restrict__ alpha_logit,
                                                     bf16* __restrict__ S, int W) {
  const int i = blockIdx.x;
  const long b = blockIdx.y;
  const int t = threadIdx.x;
  __shared__ float qi[16];
  if (t < 16) qi[t] = q[((size_t)b * W + i) * 16 + t];
  __syncthreads();
  const float gate  = sigmoid_f(gate_logit[0]);
  const float alpha = sigmoid_f(alpha_logit[0]);
  bf16* Srow = S + ((size_t)b * W + i) * W;
  for (int j = t; j < W; j += 256) {
    float val = 0.f;
    if (j <= i) {
      val = gate * kbase[(size_t)i * W + j];
      const int dd = i - j;
      const float* kr = kk + ((size_t)b * W + j) * 16;
      float s = 0.f;
#pragma unroll
      for (int r = 0; r < 16; r++) s += qi[r] * pw[r * W + dd] * kr[r];
      val += alpha * s;
    }
    Srow[j] = (bf16)val;
  }
}

// ---------------------------------------------------------------------------
// f32 (Krows x Ncols, row-major) -> bf16 transposed [Ncols][Krows].
// ---------------------------------------------------------------------------
__global__ __launch_bounds__(256) void transpose_cvt_kernel(const float* __restrict__ X,
                                                            bf16* __restrict__ Xt,
                                                            int Krows, int Ncols,
                                                            long sX, long sXt) {
  __shared__ float tile[32][33];
  const int t = threadIdx.x;
  const int tx = t & 31, ty = t >> 5;  // ty 0..7
  const long b = blockIdx.z;
  const int kb0 = blockIdx.y * 32, nb0 = blockIdx.x * 32;
  const float* Xb = X + b * sX;
  bf16* Xtb = Xt + b * sXt;
#pragma unroll
  for (int s = 0; s < 4; s++) {
    const int r = ty + s * 8;
    tile[r][tx] = Xb[(size_t)(kb0 + r) * Ncols + nb0 + tx];
  }
  __syncthreads();
#pragma unroll
  for (int s = 0; s < 4; s++) {
    const int r = ty + s * 8;  // local n
    Xtb[(size_t)(nb0 + r) * Krows + kb0 + tx] = (bf16)tile[tx][r];
  }
}

// ---------------------------------------------------------------------------
// bf16 WMMA GEMM: C = A(MxK,row) * Bt(NxK,row)^T, f32 accumulate.
// 256 thr = 8 waves; block tile 128x128x32; wave tile 64x32 = 4x2 WMMA frags.
// Fused epilogue: +bias[col], GELU(erf), +residual, store f32 or bf16.
// M,N multiples of 128; K multiple of 32 (true for all call sites).
// ---------------------------------------------------------------------------
#define BM 128
#define BN 128
#define BKK 32
#define KP 40  // padded LDS K-stride (bf16 units); row stride 80B, 16B aligned

template <bool BF16OUT, bool BIAS, bool RES, bool GELU>
__global__ __launch_bounds__(256) void gemm_bf16_kernel(const bf16* __restrict__ A,
                                                        const bf16* __restrict__ Bt,
                                                        void* __restrict__ Cv,
                                                        const float* __restrict__ bias,
                                                        const float* __restrict__ res,
                                                        int M, int N, int K,
                                                        long sA_, long sB_, long sC_) {
  __shared__ bf16 shA[BM * KP];
  __shared__ bf16 shB[BN * KP];
  const int t = threadIdx.x;
  const int lane = t & 31;
  const int wid = t >> 5;
  const int wm = wid >> 2;   // 0..1
  const int wn = wid & 3;    // 0..3
  const int m0 = blockIdx.y * BM, n0 = blockIdx.x * BN;
  const long bz = blockIdx.z;
  const bf16* gA = A + bz * sA_ + (size_t)m0 * K;
  const bf16* gB = Bt + bz * sB_ + (size_t)n0 * K;

  v8f acc[4][2];
#pragma unroll
  for (int i = 0; i < 4; i++)
#pragma unroll
    for (int j = 0; j < 2; j++)
#pragma unroll
      for (int e = 0; e < 8; e++) acc[i][j][e] = 0.0f;

  const int lr = t >> 2;        // 0..63
  const int lk = (t & 3) * 8;   // 0,8,16,24

  const int mlane = lane & 15;
  const int koffA = (lane & 16) ? 16 : 0;   // A frag: contiguous 16-K run per half
  const int koffB = (lane & 16) ? 16 : 0;   // B frag: contiguous 16-K run per half

  for (int k0 = 0; k0 < K; k0 += BKK) {
    __syncthreads();
    *(uint4*)&shA[lr * KP + lk]        = *(const uint4*)&gA[(size_t)lr * K + k0 + lk];
    *(uint4*)&shA[(lr + 64) * KP + lk] = *(const uint4*)&gA[(size_t)(lr + 64) * K + k0 + lk];
    *(uint4*)&shB[lr * KP + lk]        = *(const uint4*)&gB[(size_t)lr * K + k0 + lk];
    *(uint4*)&shB[(lr + 64) * KP + lk] = *(const uint4*)&gB[(size_t)(lr + 64) * K + k0 + lk];
    __syncthreads();

    union Frag { v16bf v; uint4 u[2]; };
    Frag fb[2];
#pragma unroll
    for (int j = 0; j < 2; j++) {
      const int n = wn * 32 + j * 16 + mlane;
      fb[j].u[0] = *(const uint4*)&shB[n * KP + koffB];
      fb[j].u[1] = *(const uint4*)&shB[n * KP + koffB + 8];
    }
#pragma unroll
    for (int i = 0; i < 4; i++) {
      Frag fa;
      const int m = wm * 64 + i * 16 + mlane;
      fa.u[0] = *(const uint4*)&shA[m * KP + koffA];
      fa.u[1] = *(const uint4*)&shA[m * KP + koffA + 8];
#pragma unroll
      for (int j = 0; j < 2; j++) {
        acc[i][j] = __builtin_amdgcn_wmma_f32_16x16x32_bf16(
            false, fa.v, false, fb[j].v, (short)0, acc[i][j], false, false);
      }
    }
  }

  // Epilogue. C/D layout: lanes 0-15 -> M=v, lanes 16-31 -> M=v+8; N = lane&15.
  const int nlane = lane & 15;
  const int mofs = (lane >> 4) * 8;
#pragma unroll
  for (int i = 0; i < 4; i++) {
#pragma unroll
    for (int j = 0; j < 2; j++) {
      const int col = n0 + wn * 32 + j * 16 + nlane;
#pragma unroll
      for (int vv = 0; vv < 8; vv++) {
        const int row = m0 + wm * 64 + i * 16 + mofs + vv;
        float c = acc[i][j][vv];
        if (BIAS) c += bias[col];
        if (GELU) c = 0.5f * c * (1.0f + erff(c * 0.70710678118654752f));
        const long gro = bz * sC_ + (long)row * N + col;
        if (RES) c += res[gro];
        if (BF16OUT) ((bf16*)Cv)[gro] = (bf16)c;
        else         ((float*)Cv)[gro] = c;
      }
    }
  }
}

// ---------------------------------------------------------------------------
extern "C" void kernel_launch(void* const* d_in, const int* in_sizes, int n_in,
                              void* d_out, int out_size, void* d_ws, size_t ws_size,
                              hipStream_t stream) {
  (void)n_in; (void)out_size; (void)ws_size;
  const float* h    = (const float*)d_in[0];
  const float* dl   = (const float*)d_in[1];
  const float* kb   = (const float*)d_in[2];
  const float* gl   = (const float*)d_in[3];
  const float* u    = (const float*)d_in[4];
  const float* v    = (const float*)d_in[5];
  const float* al   = (const float*)d_in[6];
  const float* pw_w = (const float*)d_in[7];
  const float* pb   = (const float*)d_in[8];
  const float* n1   = (const float*)d_in[9];
  const float* n2   = (const float*)d_in[10];
  const float* upw  = (const float*)d_in[11];
  const float* upb  = (const float*)d_in[12];
  const float* dww  = (const float*)d_in[13];
  const float* dwb  = (const float*)d_in[14];

  const int R = in_sizes[1];
  const int D = in_sizes[8];
  int W = 1;
  while ((long)W * W < (long)in_sizes[2]) W <<= 1;  // W*W == in_sizes[2]
  const int B = (int)((long)in_sizes[0] / ((long)W * D));
  const long BW = (long)B * W;
  const int D2 = 2 * D;

  char* p = (char*)d_ws;
  auto alloc = [&](size_t bytes) -> void* {
    void* r = (void*)p;
    p += (bytes + 255) & ~(size_t)255;
    return r;
  };

  float* hn    = (float*)alloc(sizeof(float) * BW * D);            // rmsnorm1 out
  bf16*  hnT   = (bf16*) alloc(sizeof(bf16) * (size_t)B * D * W);  // hn^T per batch
  float* qv    = (float*)alloc(sizeof(float) * BW * R);
  float* kv    = (float*)alloc(sizeof(float) * BW * R);
  float* pwt   = (float*)alloc(sizeof(float) * (size_t)R * W);     // gamma^d table
  bf16*  S     = (bf16*) alloc(sizeof(bf16) * (size_t)B * W * W);  // fused scores
  bf16*  out1  = (bf16*) alloc(sizeof(bf16) * BW * D);             // S@hn (pre-proj)
  bf16*  projT = (bf16*) alloc(sizeof(bf16) * (size_t)D * D);
  float* h2    = (float*)alloc(sizeof(float) * BW * D);
  bf16*  h2n   = (bf16*) alloc(sizeof(bf16) * BW * D);
  bf16*  upT   = (bf16*) alloc(sizeof(bf16) * (size_t)D2 * D);
  bf16*  midb  = (bf16*) alloc(sizeof(bf16) * BW * D2);
  bf16*  downT = (bf16*) alloc(sizeof(bf16) * (size_t)D * D2);

  // 1) h_norm = rmsnorm(h, norm1_scale)  (f32)
  rmsnorm_kernel<false><<<dim3((unsigned)BW), 256, 0, stream>>>(h, n1, hn, D);
  // 2) hn^T bf16 per batch (B-operand of gemm1)
  transpose_cvt_kernel<<<dim3(D / 32, W / 32, B), 256, 0, stream>>>(
      hn, hnT, W, D, (long)W * D, (long)D * W);
  // 3) q, k = l2norm(hn@u), l2norm(hn@v)
  qk_kernel<<<dim3((unsigned)BW), 256, 0, stream>>>(hn, u, v, qv, kv, D, R);
  // 4) gamma^d table
  powtab_kernel<<<dim3((R * W + 255) / 256), 256, 0, stream>>>(dl, pwt, W, R);
  // 5) fused score matrix S (bf16, causal-masked)
  buildS_kernel<<<dim3(W, B), 256, 0, stream>>>(qv, kv, pwt, kb, gl, al, S, W);
  // 6-8) weight transposes -> bf16 [N][K]
  transpose_cvt_kernel<<<dim3(D / 32, D / 32, 1), 256, 0, stream>>>(pw_w, projT, D, D, 0, 0);
  transpose_cvt_kernel<<<dim3(D2 / 32, D / 32, 1), 256, 0, stream>>>(upw, upT, D, D2, 0, 0);
  transpose_cvt_kernel<<<dim3(D / 32, D2 / 32, 1), 256, 0, stream>>>(dww, downT, D2, D, 0, 0);
  // 9) out1[b] = S[b] @ hn[b]                (batched WMMA, bf16 out)
  gemm_bf16_kernel<true, false, false, false><<<dim3(D / 128, W / 128, B), 256, 0, stream>>>(
      S, hnT, out1, nullptr, nullptr, W, D, W, (long)W * W, (long)D * W, (long)W * D);
  // 10) h2 = h + out1 @ proj_w + proj_b     (WMMA, residual+bias, f32 out)
  gemm_bf16_kernel<false, true, true, false><<<dim3(D / 128, (unsigned)(BW / 128), 1), 256, 0, stream>>>(
      out1, projT, h2, pb, h, (int)BW, D, D, 0, 0, 0);
  // 11) h2n = rmsnorm(h2, norm2_scale)      (bf16)
  rmsnorm_kernel<true><<<dim3((unsigned)BW), 256, 0, stream>>>(h2, n2, h2n, D);
  // 12) mid = gelu(h2n @ up_w + up_b)       (WMMA, bias+GELU, bf16 out)
  gemm_bf16_kernel<true, true, false, true><<<dim3(D2 / 128, (unsigned)(BW / 128), 1), 256, 0, stream>>>(
      h2n, upT, midb, upb, nullptr, (int)BW, D2, D, 0, 0, 0);
  // 13) out = h2 + mid @ down_w + down_b    (WMMA, residual+bias, f32 -> d_out)
  gemm_bf16_kernel<false, true, true, false><<<dim3(D / 128, (unsigned)(BW / 128), 1), 256, 0, stream>>>(
      midb, downT, (float*)d_out, dwb, h2, (int)BW, D, D2, 0, 0, 0);
}
